// GAT_12678743458359
// MI455X (gfx1250) — matrix operand order repaired and checked by reading, not compiled, not run on hardware
//
#include <hip/hip_runtime.h>
#include <hip/hip_bf16.h>

typedef __attribute__((ext_vector_type(16))) _Float16 v16h;
typedef __attribute__((ext_vector_type(8)))  _Float16 v8h;
typedef __attribute__((ext_vector_type(8)))  float    v8f;

#define TILE_M 128
#define TILE_N 64
#define KSTEP  32
#define KMAX   256
#define BPITCH (KMAX + 8)   // 264 halves = 528 B row pitch: 16B aligned, 4-bank skew

// ---------------------------------------------------------------------------
// Mixed-precision GEMM: D[M,Ncols] = A[M,K] * B[K,Ncols]; f32 in, f16 WMMA,
// f32 accumulate. 256 threads = 8 wave32; each wave computes a 16x64 strip.
// Whole KxTILE_N B-panel staged in LDS once; K loop has no barriers.
// ---------------------------------------------------------------------------
__global__ __launch_bounds__(256) void gat_gemm_wmma(
    const float* __restrict__ A, const float* __restrict__ B,
    float* __restrict__ D, int M, int K, int Ncols)
{
    __shared__ _Float16 sBt[TILE_N][BPITCH];     // [n][k], f16

    const int tid  = threadIdx.x;
    const int wave = tid >> 5;
    const int lane = tid & 31;
    const int m0   = blockIdx.x * TILE_M + wave * 16;
    const int n0   = blockIdx.y * TILE_N;

    // ---- stage full B panel: K rows x 64 cols, f32 -> f16, transposed ----
    {
        const int tn = (tid & 15) * 4;           // col quad within tile
        const int tk = tid >> 4;                 // k row 0..15
        if (n0 + TILE_N <= Ncols) {              // full tile: vector fast path
            for (int k = tk; k < K; k += 16) {
                const float4 v = *(const float4*)(B + k * Ncols + n0 + tn);
                sBt[tn + 0][k] = (_Float16)v.x;
                sBt[tn + 1][k] = (_Float16)v.y;
                sBt[tn + 2][k] = (_Float16)v.z;
                sBt[tn + 3][k] = (_Float16)v.w;
            }
        } else {                                 // ragged tile (layer 3: Ncols=5)
            for (int k = tk; k < K; k += 16)
                #pragma unroll
                for (int j = 0; j < 4; ++j) {
                    int ng = n0 + tn + j;
                    float v = (ng < Ncols) ? B[k * Ncols + ng] : 0.0f;
                    sBt[tn + j][k] = (_Float16)v;
                }
        }
    }
    __syncthreads();

    const int mrow   = m0 + (lane & 15);
    const int mclamp = (mrow < M) ? mrow : (M - 1);  // EXEC stays all-1s for WMMA
    const int kbaseA = (lane >> 4) * 8;              // A fragment K sub-base
    const int kbaseB = (lane >> 4) * 8;              // B fragment K sub-base
    const float* arow0 = A + mclamp * K;

    v8f acc[4];
    #pragma unroll
    for (int t = 0; t < 4; ++t)
        #pragma unroll
        for (int v = 0; v < 8; ++v) acc[t][v] = 0.0f;

    #pragma unroll 4
    for (int k0 = 0; k0 < K; k0 += KSTEP) {
        // A fragment straight from global (two 32B loads, then cvt)
        const float* arow = arow0 + k0;
        v8f a_lo = *(const v8f*)(arow + kbaseA);
        v8f a_hi = *(const v8f*)(arow + 16 + kbaseA);
        if (k0 + KSTEP < K)                           // global_prefetch_b8
            __builtin_prefetch(arow + KSTEP, 0, 1);
        v8h al16 = __builtin_convertvector(a_lo, v8h);
        v8h ah16 = __builtin_convertvector(a_hi, v8h);
        v16h afrag = __builtin_shufflevector(al16, ah16,
            0,1,2,3,4,5,6,7,8,9,10,11,12,13,14,15);

        const int kb = k0 + kbaseB;
        #pragma unroll
        for (int t = 0; t < 4; ++t) {
            int n = t * 16 + (lane & 15);
            v8h b_lo = *(const v8h*)&sBt[n][kb];
            v8h b_hi = *(const v8h*)&sBt[n][kb + 16];
            v16h bfrag = __builtin_shufflevector(b_lo, b_hi,
                0,1,2,3,4,5,6,7,8,9,10,11,12,13,14,15);
            acc[t] = __builtin_amdgcn_wmma_f32_16x16x32_f16(
                false, afrag, false, bfrag, (short)0, acc[t], false, false);
        }
    }

    // store D (C/D layout: VGPR v -> M = v + 8*(lane>=16); N = lane&15)
    #pragma unroll
    for (int t = 0; t < 4; ++t) {
        int col = n0 + t * 16 + (lane & 15);
        if (col < Ncols) {
            #pragma unroll
            for (int v = 0; v < 8; ++v) {
                int row = m0 + v + ((lane >> 4) << 3);
                if (row < M) D[row * Ncols + col] = acc[t][v];
            }
        }
    }
}

// ---------------------------------------------------------------------------
// Attention coefficients: al_s/al_d[n,h] = sum_c h[n,h,c]*a_{src,dst}[h,c]
// ---------------------------------------------------------------------------
__global__ void gat_attn_coef(const float* __restrict__ h,
    const float* __restrict__ a_src, const float* __restrict__ a_dst,
    float* __restrict__ al_s, float* __restrict__ al_d,
    int n_nodes, int heads, int C)
{
    int idx = blockIdx.x * blockDim.x + threadIdx.x;
    if (idx >= n_nodes * heads) return;
    int node = idx / heads, hd = idx % heads;
    const float* hp = h + node * heads * C + hd * C;
    const float* as = a_src + hd * C;
    const float* ad = a_dst + hd * C;
    float ss = 0.f, sd = 0.f;
    for (int c = 0; c < C; ++c) { float v = hp[c]; ss += v * as[c]; sd += v * ad[c]; }
    al_s[idx] = ss; al_d[idx] = sd;
}

// order-preserving float<->uint map (atomicMax on uint == float max, no CAS)
__device__ __forceinline__ unsigned f2key(float f) {
    unsigned u = __float_as_uint(f);
    return (u & 0x80000000u) ? ~u : (u | 0x80000000u);
}
__device__ __forceinline__ float key2f(unsigned k) {
    return __uint_as_float((k & 0x80000000u) ? (k & 0x7FFFFFFFu) : ~k);
}
__device__ __forceinline__ void edge_nodes(const int* __restrict__ ei, int E,
                                           int e, int& s, int& d) {
    if (e < E) { s = ei[e]; d = ei[E + e]; }
    else       { s = e - E; d = e - E; }          // implicit self-loops
}

// ---------------------------------------------------------------------------
// Pass 1: segment max of leaky-relu logits (uint-encoded atomicMax, L2-resident)
// ---------------------------------------------------------------------------
__global__ void gat_edge_max(const int* __restrict__ ei, int E, int Etot,
    const float* __restrict__ al_s, const float* __restrict__ al_d,
    unsigned* __restrict__ m_enc, int heads)
{
    int e = blockIdx.x * blockDim.x + threadIdx.x;
    if (e >= Etot) return;
    int s, d; edge_nodes(ei, E, e, s, d);
    for (int hd = 0; hd < heads; ++hd) {
        float v = al_s[s * heads + hd] + al_d[d * heads + hd];
        v = (v > 0.f) ? v : 0.2f * v;
        atomicMax(&m_enc[d * heads + hd], f2key(v));
    }
}

// ---------------------------------------------------------------------------
// Pass 2 (fused): p = exp(e - m[d]); ssum[d] += p; out[d,:] += p * h[s,:]
// One wave per edge; h/out stay L2-resident (51 MB each << 192 MB L2).
// ---------------------------------------------------------------------------
__global__ __launch_bounds__(256) void gat_edge_agg(const int* __restrict__ ei,
    int E, int Etot,
    const float* __restrict__ al_s, const float* __restrict__ al_d,
    const unsigned* __restrict__ m_enc,
    const float* __restrict__ h, float* __restrict__ out,
    float* __restrict__ ssum, int heads, int C)
{
    int e    = (blockIdx.x * blockDim.x + threadIdx.x) >> 5;
    int lane = threadIdx.x & 31;
    if (e >= Etot) return;
    int s, d; edge_nodes(ei, E, e, s, d);

    float p[4];                                   // heads <= 4
    for (int hd = 0; hd < heads; ++hd) {
        float v = al_s[s * heads + hd] + al_d[d * heads + hd];
        v = (v > 0.f) ? v : 0.2f * v;
        p[hd] = __expf(v - key2f(m_enc[d * heads + hd]));
    }
    if (lane < heads) atomicAdd(&ssum[d * heads + lane], p[lane]);

    int HC = heads * C;
    const float* hs = h + s * HC;
    float* od = out + d * HC;
    for (int f = lane; f < HC; f += 32)
        atomicAdd(&od[f], p[f / C] * hs[f]);
}

// ---------------------------------------------------------------------------
// Pass 3: normalize by ssum, add bias, optional ReLU (in place)
// ---------------------------------------------------------------------------
__global__ void gat_finalize(float* __restrict__ out, const float* __restrict__ ssum,
    const float* __restrict__ b, int n_nodes, int heads, int C, int relu)
{
    int HC  = heads * C;
    int idx = blockIdx.x * blockDim.x + threadIdx.x;
    if (idx >= n_nodes * HC) return;
    int node = idx / HC, f = idx % HC;
    float v = out[idx] / (ssum[node * heads + f / C] + 1e-16f) + b[f];
    if (relu) v = (v > 0.f) ? v : 0.f;
    out[idx] = v;
}

// ---------------------------------------------------------------------------
static void run_layer(const float* xin, int K, const float* W,
                      const float* a_s, const float* a_d, const float* b,
                      int heads, int C, int relu,
                      float* hbuf, float* outbuf,
                      float* al_s, float* al_d, unsigned* m_enc, float* ssum,
                      const int* ei, int N, int E, int Etot, hipStream_t stream)
{
    int HC = heads * C;
    dim3 gg((N + TILE_M - 1) / TILE_M, (HC + TILE_N - 1) / TILE_N);
    gat_gemm_wmma<<<gg, 256, 0, stream>>>(xin, W, hbuf, N, K, HC);

    int nh = N * heads;
    gat_attn_coef<<<(nh + 255) / 256, 256, 0, stream>>>(hbuf, a_s, a_d,
                                                        al_s, al_d, N, heads, C);
    hipMemsetAsync(m_enc, 0, (size_t)nh * 4, stream);        // 0 < key(-inf)
    hipMemsetAsync(ssum,  0, (size_t)nh * 4, stream);
    hipMemsetAsync(outbuf, 0, (size_t)N * HC * 4, stream);

    gat_edge_max<<<(Etot + 255) / 256, 256, 0, stream>>>(ei, E, Etot,
                                                         al_s, al_d, m_enc, heads);
    gat_edge_agg<<<(Etot + 7) / 8, 256, 0, stream>>>(ei, E, Etot, al_s, al_d,
                                                     m_enc, hbuf, outbuf, ssum,
                                                     heads, C);
    int tot = N * HC;
    gat_finalize<<<(tot + 255) / 256, 256, 0, stream>>>(outbuf, ssum, b,
                                                        N, heads, C, relu);
}

extern "C" void kernel_launch(void* const* d_in, const int* in_sizes, int n_in,
                              void* d_out, int out_size, void* d_ws, size_t ws_size,
                              hipStream_t stream)
{
    const float* x      = (const float*)d_in[0];
    const int*   ei     = (const int*)d_in[1];
    const float* W1     = (const float*)d_in[2];
    const float* a_src1 = (const float*)d_in[3];
    const float* a_dst1 = (const float*)d_in[4];
    const float* b1     = (const float*)d_in[5];
    const float* W2     = (const float*)d_in[6];
    const float* a_src2 = (const float*)d_in[7];
    const float* a_dst2 = (const float*)d_in[8];
    const float* b2     = (const float*)d_in[9];
    const float* W3     = (const float*)d_in[10];
    const float* a_src3 = (const float*)d_in[11];
    const float* a_dst3 = (const float*)d_in[12];
    const float* b3     = (const float*)d_in[13];

    const int F_IN  = 128, HEADS = 4, HID = 64, NCLS = 5;
    const int N     = in_sizes[0] / F_IN;     // 50000
    const int E     = in_sizes[1] / 2;        // 800000
    const int Etot  = E + N;                  // + self loops
    const int HC    = HEADS * HID;            // 256

    // workspace carve-out (~106 MB)
    float*    bufA  = (float*)d_ws;                         // GEMM output h
    float*    bufB  = bufA + (size_t)N * HC;                // aggregation out / next input
    float*    al_s  = bufB + (size_t)N * HC;
    float*    al_d  = al_s + (size_t)N * HEADS;
    unsigned* m_enc = (unsigned*)(al_d + (size_t)N * HEADS);
    float*    ssum  = (float*)(m_enc + (size_t)N * HEADS);
    (void)ws_size; (void)n_in; (void)out_size;

    // L1: x[ N,128 ] -> bufB[ N,256 ] (ReLU)
    run_layer(x,    F_IN, W1, a_src1, a_dst1, b1, HEADS, HID, 1,
              bufA, bufB, al_s, al_d, m_enc, ssum, ei, N, E, Etot, stream);
    // L2: bufB[ N,256 ] -> bufB[ N,256 ] (ReLU); bufB free after GEMM, reused as out
    run_layer(bufB, HC,   W2, a_src2, a_dst2, b2, HEADS, HID, 1,
              bufA, bufB, al_s, al_d, m_enc, ssum, ei, N, E, Etot, stream);
    // L3: bufB[ N,256 ] -> d_out[ N,5 ] (no ReLU, heads=1)
    run_layer(bufB, HC,   W3, a_src3, a_dst3, b3, 1, NCLS, 0,
              bufA, (float*)d_out, al_s, al_d, m_enc, ssum, ei, N, E, Etot, stream);
}
